// WindowAttention_33904471835149
// MI455X (gfx1250) — matrix operand order repaired and checked by reading, not compiled, not run on hardware
//
#include <hip/hip_runtime.h>

typedef __bf16 bf16_t;
typedef __attribute__((ext_vector_type(16))) __bf16 v16bf;
typedef __attribute__((ext_vector_type(8)))  __bf16 v8bf;
typedef __attribute__((ext_vector_type(4)))  __bf16 v4bf;
typedef __attribute__((ext_vector_type(8)))  float  v8f;
typedef __attribute__((ext_vector_type(4)))  float  v4f;

#define WSZ   7
#define NWIN  49
#define NPAD  64
#define HEADS 12
#define DH    32
#define CC    384
#define CC3   1152
#define HN    8
#define IMG   56
#define QSCALE 0.17677669529663687f  /* 32^-0.5 */

// ---------------------------------------------------------------------------
// WMMA helper: D = A(16x32 bf16) x B(32x16 bf16) + C(16x16 f32), wave32
// ---------------------------------------------------------------------------
__device__ __forceinline__ v8f wmma_bf16(v16bf a, v16bf b, v8f c) {
  return __builtin_amdgcn_wmma_f32_16x16x32_bf16(false, a, false, b,
                                                 (short)0, c, false, false);
}

// A fragment from row-major S[row][ld] (bf16), same K order as B fragments:
// lane L holds row M = L%16; lanes<16 take K = kbase+[0..15],
// lanes>=16 take K = kbase+16+[0..15].
__device__ __forceinline__ v16bf load_a_frag(const bf16_t* S, int ld, int mbase,
                                             int kbase, int l16, int half) {
  return *(const v16bf*)(S + (size_t)(mbase + l16) * ld + kbase + half * 16);
}

// B fragment: B[K][N] = S[N][K] (S row-major, K contiguous per lane).
// lanes<16 hold K=kbase+[0..15], lanes>=16 hold K=kbase+16+[0..15]; N = lane%16.
__device__ __forceinline__ v16bf load_b_frag(const bf16_t* S, int ld, int nbase,
                                             int kbase, int l16, int half) {
  return *(const v16bf*)(S + (size_t)(nbase + l16) * ld + kbase + half * 16);
}

// ---------------------------------------------------------------------------
// Prep: f32 weights -> bf16, expand relative-position bias to [12][64][64] f32
// ---------------------------------------------------------------------------
__global__ void prep_kernel(const float* __restrict__ wqkv_f,
                            const float* __restrict__ wout_f,
                            const float* __restrict__ rel,
                            bf16_t* __restrict__ wq_b,
                            bf16_t* __restrict__ wo_b,
                            float* __restrict__ bias) {
  int i = blockIdx.x * 256 + threadIdx.x;
  const int N1 = CC3 * CC;
  const int N2 = CC * CC;
  const int N3 = HEADS * NPAD * NPAD;
  if (i < N1) {
    wq_b[i] = (bf16_t)wqkv_f[i];
  } else if (i < N1 + N2) {
    wo_b[i - N1] = (bf16_t)wout_f[i - N1];
  } else if (i < N1 + N2 + N3) {
    int t = i - N1 - N2;
    int h = t >> 12;
    int r = (t >> 6) & 63;
    int c = t & 63;
    float v = 0.f;
    if (r < NWIN && c < NWIN) {
      int yi = r / WSZ, xi = r % WSZ;
      int yj = c / WSZ, xj = c % WSZ;
      v = rel[h * 169 + (yi - yj + 6) * 13 + (xi - xj + 6)];
    }
    bias[t] = v;
  }
}

// ---------------------------------------------------------------------------
// Fused window attention: one block = one 7x7 (dilated) window, all 12 heads
// + output projection. 256 threads = 8 wave32.
// ---------------------------------------------------------------------------
__global__ __launch_bounds__(256)
void win_attn_fused(const float* __restrict__ x,
                    const float* __restrict__ b_qkv,
                    const float* __restrict__ b_out,
                    const bf16_t* __restrict__ wqkv,
                    const bf16_t* __restrict__ wout,
                    const float* __restrict__ bias,
                    float* __restrict__ out) {
  __shared__ bf16_t sXW[NPAD * CC];     // window activations, bf16, rows 49..63 = 0
  __shared__ bf16_t sQ[NPAD * DH];      // q (scaled), row-major
  __shared__ bf16_t sK[NPAD * DH];      // k, row-major [key][d]
  __shared__ bf16_t sVT[DH * NPAD];     // v transposed [d][key]
  __shared__ bf16_t sATT[NPAD * NPAD];  // unnormalized exp(sim - max)
  __shared__ bf16_t sAOUT[NPAD * CC];   // concat head outputs (pre out-proj)
  __shared__ float  sRedMax[NPAD * 2];  // stage2a: per-wave-pair row max; later: 1/rowsum
  __shared__ float  sRedSum[NPAD * 2];

  const int tid  = threadIdx.x;
  const int lane = tid & 31;
  const int wv   = tid >> 5;
  const int l16  = lane & 15;
  const int half = (lane >> 4) & 1;

  const int win = blockIdx.x;
  const int bi  = win >> 6;
  const int hh  = (win >> 3) & 7;
  const int ww  = win & 7;

  // hint L2 on the shared weight tables
  __builtin_prefetch(wqkv, 0, 1);
  __builtin_prefetch(wout, 0, 1);

  // ---- stage 0: gather window pixels (read-once -> non-temporal), f32->bf16 ----
  for (int i4 = tid; i4 < (NPAD * CC) / 4; i4 += 256) {
    int e  = i4 * 4;
    int r  = e / CC;
    int c0 = e % CC;
    v4f v = {};
    if (r < NWIN) {
      int p = r / WSZ, q = r % WSZ;
      int Hy = p * HN + hh, Wx = q * HN + ww;
      v = __builtin_nontemporal_load(
            (const v4f*)(x + (((size_t)bi * IMG + Hy) * IMG + Wx) * CC + c0));
    }
    v4bf pk;
    pk[0] = (bf16_t)v[0]; pk[1] = (bf16_t)v[1];
    pk[2] = (bf16_t)v[2]; pk[3] = (bf16_t)v[3];
    *(v4bf*)(sXW + e) = pk;
  }
  __syncthreads();

  for (int head = 0; head < HEADS; ++head) {
    // ---- stage 1: q/k/v = xw @ W^T + b for this head (24 tiles, 12 k-steps) ----
    for (int t = wv; t < 24; t += 8) {
      int Mt     = t / 6;
      int ct     = t % 6;
      int sel    = ct >> 1;            // 0=q 1=k 2=v
      int coloff = (ct & 1) * 16;
      int jrow   = sel * CC + head * DH + coloff + l16;
      const bf16_t* bp = wqkv + (size_t)jrow * CC;  // B[K][N] = wqkv[jrow(N)][K]
      v8f acc = {};
      #pragma unroll
      for (int kk = 0; kk < 12; ++kk) {
        v16bf a = load_a_frag(sXW, CC, Mt * 16, kk * 32, l16, half);
        v16bf b = *(const v16bf*)(bp + kk * 32 + half * 16);
        acc = wmma_bf16(a, b, acc);
      }
      float bj   = b_qkv[jrow];
      int   colh = coloff + l16;       // 0..31 within head
      int   r0   = Mt * 16 + half * 8; // D rows: r0..r0+7
      if (sel == 0) {
        #pragma unroll
        for (int e = 0; e < 8; ++e)
          sQ[(r0 + e) * DH + colh] = (bf16_t)((acc[e] + bj) * QSCALE);
      } else if (sel == 1) {
        #pragma unroll
        for (int e = 0; e < 8; ++e)
          sK[(r0 + e) * DH + colh] = (bf16_t)(acc[e] + bj);
      } else {
        v8bf pk;
        #pragma unroll
        for (int e = 0; e < 8; ++e) pk[e] = (bf16_t)(acc[e] + bj);
        *(v8bf*)(sVT + (size_t)colh * NPAD + r0) = pk;  // contiguous along key
      }
    }
    __syncthreads();

    // ---- stage 2: sim = q k^T + bias; masked softmax stats ----
    {
      int Mt  = wv >> 1;
      int ntb = (wv & 1) * 2;
      v16bf aq = load_a_frag(sQ, DH, Mt * 16, 0, l16, half);
      v16bf b0 = load_b_frag(sK, DH, (ntb + 0) * 16, 0, l16, half);
      v16bf b1 = load_b_frag(sK, DH, (ntb + 1) * 16, 0, l16, half);
      v8f z = {};
      v8f s0 = wmma_bf16(aq, b0, z);
      v8f s1 = wmma_bf16(aq, b1, z);
      int col0 = (ntb + 0) * 16 + l16;
      int col1 = (ntb + 1) * 16 + l16;
      const float* bg = bias + (size_t)head * NPAD * NPAD;
      int r0 = Mt * 16 + half * 8;
      #pragma unroll
      for (int e = 0; e < 8; ++e) {
        int r = r0 + e;
        s0[e] = (col0 < NWIN) ? (s0[e] + bg[r * NPAD + col0]) : -1e30f;
        s1[e] = (col1 < NWIN) ? (s1[e] + bg[r * NPAD + col1]) : -1e30f;
      }
      // partial row max over this wave's two column tiles
      #pragma unroll
      for (int e = 0; e < 8; ++e) {
        float m = fmaxf(s0[e], s1[e]);
        #pragma unroll
        for (int d = 1; d < 16; d <<= 1) m = fmaxf(m, __shfl_xor(m, d, 16));
        if (l16 == 0) sRedMax[(r0 + e) * 2 + (wv & 1)] = m;
      }
      __syncthreads();
      #pragma unroll
      for (int e = 0; e < 8; ++e) {
        int r = r0 + e;
        float m  = fmaxf(sRedMax[r * 2], sRedMax[r * 2 + 1]);
        float p0 = (col0 < NWIN) ? __expf(s0[e] - m) : 0.f;
        float p1 = (col1 < NWIN) ? __expf(s1[e] - m) : 0.f;
        sATT[r * NPAD + col0] = (bf16_t)p0;
        sATT[r * NPAD + col1] = (bf16_t)p1;
        float su = p0 + p1;
        #pragma unroll
        for (int d = 1; d < 16; d <<= 1) su += __shfl_xor(su, d, 16);
        if (l16 == 0) sRedSum[r * 2 + (wv & 1)] = su;
      }
    }
    __syncthreads();

    // ---- stage 2c: one v_rcp_f32 per row, shared by everyone (reuses sRedMax) ----
    if (tid < NPAD)
      sRedMax[tid] = __builtin_amdgcn_rcpf(sRedSum[tid * 2] + sRedSum[tid * 2 + 1]);
    __syncthreads();

    // ---- stage 3: out_head = (exp-sim @ v) * (1/rowsum) ----
    {
      int Mt = wv >> 1;
      int Nt = wv & 1;
      v8f o = {};
      #pragma unroll
      for (int kk = 0; kk < 2; ++kk) {
        v16bf a = load_a_frag(sATT, NPAD, Mt * 16, kk * 32, l16, half);
        v16bf b = load_b_frag(sVT, NPAD, Nt * 16, kk * 32, l16, half);
        o = wmma_bf16(a, b, o);
      }
      int col = head * DH + Nt * 16 + l16;
      int r0  = Mt * 16 + half * 8;
      #pragma unroll
      for (int e = 0; e < 8; ++e) {
        int r = r0 + e;
        sAOUT[(size_t)r * CC + col] = (bf16_t)(o[e] * sRedMax[r]);
      }
    }
    __syncthreads();
  }

  // ---- stage 4: out = aout @ w_out^T + b_out, scatter with inverse window perm ----
  for (int t = wv; t < 96; t += 8) {
    int Mt = t / 24;
    int Nt = t % 24;
    const bf16_t* bp = wout + (size_t)(Nt * 16 + l16) * CC;  // B[K][N] = wout[N][K]
    v8f acc = {};
    #pragma unroll
    for (int kk = 0; kk < 12; ++kk) {
      v16bf a = load_a_frag(sAOUT, CC, Mt * 16, kk * 32, l16, half);
      v16bf b = *(const v16bf*)(bp + kk * 32 + half * 16);
      acc = wmma_bf16(a, b, acc);
    }
    int col  = Nt * 16 + l16;
    float bb = b_out[col];
    int r0   = Mt * 16 + half * 8;
    #pragma unroll
    for (int e = 0; e < 8; ++e) {
      int r = r0 + e;
      if (r < NWIN) {
        int p = r / WSZ, q = r % WSZ;
        int Hy = p * HN + hh, Wx = q * HN + ww;
        __builtin_nontemporal_store(
            acc[e] + bb,
            out + (((size_t)bi * IMG + Hy) * IMG + Wx) * CC + col);
      }
    }
  }
}

// ---------------------------------------------------------------------------
extern "C" void kernel_launch(void* const* d_in, const int* in_sizes, int n_in,
                              void* d_out, int out_size, void* d_ws, size_t ws_size,
                              hipStream_t stream) {
  const float* x      = (const float*)d_in[0];
  const float* wqkv_f = (const float*)d_in[1];
  const float* bqkv   = (const float*)d_in[2];
  const float* rel    = (const float*)d_in[3];
  const float* wout_f = (const float*)d_in[4];
  const float* bout   = (const float*)d_in[5];
  float* out = (float*)d_out;

  char* ws = (char*)d_ws;
  bf16_t* wq_b = (bf16_t*)ws;                                   // 1152*384*2 B
  bf16_t* wo_b = (bf16_t*)(ws + (size_t)CC3 * CC * 2);          // 384*384*2 B
  float*  bias = (float*)(ws + (size_t)CC3 * CC * 2 + (size_t)CC * CC * 2);

  const int nprep = CC3 * CC + CC * CC + HEADS * NPAD * NPAD;
  prep_kernel<<<(nprep + 255) / 256, 256, 0, stream>>>(wqkv_f, wout_f, rel,
                                                       wq_b, wo_b, bias);

  const int nwindows = 32 * 8 * 8;  // b * hn * wn = 2048
  win_attn_fused<<<nwindows, 256, 0, stream>>>(x, bqkv, bout, wq_b, wo_b,
                                               bias, out);
}
